// BatchLoraLinear_52132313039179
// MI455X (gfx1250) — matrix operand order repaired and checked by reading, not compiled, not run on hardware
//
#include <hip/hip_runtime.h>

#define B_DIM 8
#define S_DIM 4096
#define D_DIM 2048
#define R_DIM 64

typedef __attribute__((ext_vector_type(16))) __bf16 v16bf;
typedef __attribute__((ext_vector_type(8)))  float  v8f;
typedef __attribute__((ext_vector_type(4)))  float  f32x4;   // native vector: OK for nontemporal builtins

union FragBF {
    v16bf v;
    unsigned u[8];
    uint4 q[2];
};

// sched_barrier mask: classes ALLOWED to cross.
// 0x1 non-mem | 0x2 VALU | 0x4 SALU | 0x10/0x20/0x40 VMEM  => 0x77
// Blocked: 0x8 (MFMA/WMMA), 0x80/0x100/0x200 (DS)
#define SCHED_PIN_DS_WMMA() __builtin_amdgcn_sched_barrier(0x77)

// Pack two f32 into one dword of bf16 using v_perm_b32 (1 op) after a
// round-to-nearest bias add (+0x8000) on each source.
__device__ __forceinline__ unsigned pack2_bf16(float lo, float hi) {
    unsigned ul = __float_as_uint(lo) + 0x8000u;
    unsigned uh = __float_as_uint(hi) + 0x8000u;
    return __builtin_amdgcn_perm(uh, ul, 0x07060302u);
}

// One block: 8 waves, each wave computes a 16(M) x 64(N) strip for one batch.
// Whole per-batch weight matrix staged to LDS as bf16 WMMA B-fragments.
__global__ void __launch_bounds__(256)
BatchLoraLinear_wmma_bf16(const float* __restrict__ x,
                          const float* __restrict__ w,
                          float* __restrict__ out) {
    extern __shared__ unsigned lds[];   // 64 chunks * 1024 dwords = 256 KB

    const int b    = blockIdx.x >> 5;   // 32 M-blocks per batch
    const int mblk = blockIdx.x & 31;
    const int t    = threadIdx.x;       // 0..255
    const int wave = t >> 5;
    const int lane = t & 31;

    // ---------------------------------------------------------------
    // Stage weights for batch b into LDS, pre-swizzled to B-fragment
    // layout: dword index = chunk*1024 + (nt*32 + lane)*8 + v
    //   lane -> n = nt*16 + (lane&15); K half = (lane&16)
//   dword v holds bf16{B[k][n], B[k+1][n]}, k = chunk*32 + ((lane&16)>>1) + 2v + (v>=4 ? 8 : 0)
    // ---------------------------------------------------------------
    const float* wb = w + (size_t)b * D_DIM * R_DIM;
    for (int g = t; g < 64 * 1024; g += 256) {
        const int c  = g >> 10;          // K-chunk (32 wide)
        const int f  = g & 1023;
        const int nt = f >> 8;           // N-tile 0..3
        const int l  = (f >> 3) & 31;    // lane slot
        const int v  = f & 7;            // dword within fragment
        const int n  = nt * 16 + (l & 15);
        const int k  = c * 32 + ((l & 16) >> 1) + 2 * v + ((v & 4) << 1);
        const float f0 = wb[(size_t)k * R_DIM + n];
        const float f1 = wb[(size_t)(k + 1) * R_DIM + n];
        lds[g] = pack2_bf16(f0, f1);
    }
    __syncthreads();

    // ---------------------------------------------------------------
    // Compute: wave handles rows [m0, m0+16)
    // ---------------------------------------------------------------
    const int m0   = mblk * 128 + wave * 16;
    const int mrow = m0 + (lane & 15);
    const float* xrow = x + ((size_t)b * S_DIM + mrow) * D_DIM;
    const int ksub = (lane & 16) >> 1;   // 0 for lanes 0-15, 8 for 16-31

    v8f acc0 = {}, acc1 = {}, acc2 = {}, acc3 = {};

    const uint4* lds4  = reinterpret_cast<const uint4*>(lds);
    const uint4* lbase = lds4 + lane * 2;

    // --- B-fragment pipeline prologue: chunk 0 ---
    uint4 nb0 = lbase[0],   nb1 = lbase[1];
    uint4 nb2 = lbase[64],  nb3 = lbase[65];
    uint4 nb4 = lbase[128], nb5 = lbase[129];
    uint4 nb6 = lbase[192], nb7 = lbase[193];

    #pragma unroll 2
    for (int c = 0; c < 64; ++c) {
        const int kt = c * 32;

        // A fragment (16-bit A 16x32 layout): VGPR0-3 = K kb..kb+7,
        // VGPR4-7 = K kb+16..kb+23, kb = kt + (lane>=16 ? 8 : 0)
        const f32x4* pa = reinterpret_cast<const f32x4*>(xrow + kt + ksub);
        const f32x4 a0 = __builtin_nontemporal_load(pa + 0);
        const f32x4 a1 = __builtin_nontemporal_load(pa + 1);
        const f32x4 a2 = __builtin_nontemporal_load(pa + 4);  // +16 floats
        const f32x4 a3 = __builtin_nontemporal_load(pa + 5);

        FragBF af;
        af.u[0] = pack2_bf16(a0.x, a0.y);
        af.u[1] = pack2_bf16(a0.z, a0.w);
        af.u[2] = pack2_bf16(a1.x, a1.y);
        af.u[3] = pack2_bf16(a1.z, a1.w);
        af.u[4] = pack2_bf16(a2.x, a2.y);
        af.u[5] = pack2_bf16(a2.z, a2.w);
        af.u[6] = pack2_bf16(a3.x, a3.y);
        af.u[7] = pack2_bf16(a3.z, a3.w);

        // Current chunk's B fragments (prefetched last iteration)
        FragBF bf0, bf1, bf2, bf3;
        bf0.q[0] = nb0; bf0.q[1] = nb1;
        bf1.q[0] = nb2; bf1.q[1] = nb3;
        bf2.q[0] = nb4; bf2.q[1] = nb5;
        bf3.q[0] = nb6; bf3.q[1] = nb7;

        // Prefetch next chunk's B fragments (wraps on last iteration:
        // branch-free, always valid)
        const uint4* lb = lbase + (((c + 1) & 63) << 8);
        nb0 = lb[0];    nb1 = lb[1];
        nb2 = lb[64];   nb3 = lb[65];
        nb4 = lb[128];  nb5 = lb[129];
        nb6 = lb[192];  nb7 = lb[193];

        // Pin: DS prefetch above, WMMAs below; VALU/VMEM may still cross.
        SCHED_PIN_DS_WMMA();

        acc0 = __builtin_amdgcn_wmma_f32_16x16x32_bf16(
            false, af.v, false, bf0.v, (short)0, acc0, false, false);
        acc1 = __builtin_amdgcn_wmma_f32_16x16x32_bf16(
            false, af.v, false, bf1.v, (short)0, acc1, false, false);
        acc2 = __builtin_amdgcn_wmma_f32_16x16x32_bf16(
            false, af.v, false, bf2.v, (short)0, acc2, false, false);
        acc3 = __builtin_amdgcn_wmma_f32_16x16x32_bf16(
            false, af.v, false, bf3.v, (short)0, acc3, false, false);
    }

    // ---------------------------------------------------------------
    // Store C: VGPR r -> row m0 + r + (lane>=16 ? 8 : 0), col lane&15
    // (streamed once: non-temporal)
    // ---------------------------------------------------------------
    float* orow = out + ((size_t)b * S_DIM + m0 + ((lane & 16) >> 1)) * R_DIM
                      + (lane & 15);
    #pragma unroll
    for (int r = 0; r < 8; ++r) {
        __builtin_nontemporal_store(acc0[r], &orow[(size_t)r * R_DIM + 0]);
        __builtin_nontemporal_store(acc1[r], &orow[(size_t)r * R_DIM + 16]);
        __builtin_nontemporal_store(acc2[r], &orow[(size_t)r * R_DIM + 32]);
        __builtin_nontemporal_store(acc3[r], &orow[(size_t)r * R_DIM + 48]);
    }
}

extern "C" void kernel_launch(void* const* d_in, const int* in_sizes, int n_in,
                              void* d_out, int out_size, void* d_ws, size_t ws_size,
                              hipStream_t stream) {
    (void)in_sizes; (void)n_in; (void)out_size; (void)d_ws; (void)ws_size;
    const float* x = (const float*)d_in[0];
    const float* w = (const float*)d_in[1];
    float* out     = (float*)d_out;

    const dim3 grid(B_DIM * (S_DIM / 128));  // 256 blocks
    const dim3 block(256);                   // 8 waves (wave32)
    const size_t shmem = 64 * 1024 * sizeof(unsigned);  // 256 KB LDS (WGP has 320 KB)

    BatchLoraLinear_wmma_bf16<<<grid, block, shmem, stream>>>(x, w, out);
}
